// Matcher_46660524704446
// MI455X (gfx1250) — matrix operand order repaired and checked by reading, not compiled, not run on hardware
//
#include <hip/hip_runtime.h>
#include <hip/hip_bf16.h>

// ---------------------------------------------------------------------------
// Problem constants (from reference)
// ---------------------------------------------------------------------------
#define D_KEY   128
#define BANK_N  9720
#define BANK_P  9728      // padded to multiple of 32 (K pad for WMMA)
#define N_COL   1620
#define N_PAD   1664      // padded to 13 * 128 (N block tiles)
#define D_VAL   512
#define C_CH    512
#define OBJ_N   3

typedef __attribute__((ext_vector_type(16))) __bf16        v16bf;
typedef __attribute__((ext_vector_type(8)))  float         v8f;
typedef __attribute__((ext_vector_type(4)))  unsigned int  u32x4;

union FragCvt { u32x4 u[2]; v16bf v; };      // 32B bf16 fragment (8 VGPRs)
union Pack8   { u32x4 u;    __bf16 h[8]; };  // 8 packed bf16

// low 32 bits of a flat shared pointer == allocation-relative LDS byte offset
__device__ __forceinline__ unsigned lds_off(const void* p) {
    return (unsigned)(unsigned long long)p;
}

// GLOBAL_LOAD_ASYNC_TO_LDS_B128: DMA 16B/lane memory -> LDS, tracked by ASYNCcnt
__device__ __forceinline__ void async_ld_b128(unsigned ldsoff, const void* gptr) {
    asm volatile("global_load_async_to_lds_b128 %0, %1, off"
                 :: "v"(ldsoff), "v"((unsigned long long)gptr)
                 : "memory");
}

template <int CNT>
__device__ __forceinline__ void wait_asynccnt() {
#if __has_builtin(__builtin_amdgcn_s_wait_asynccnt)
    __builtin_amdgcn_s_wait_asynccnt(CNT);
#else
    asm volatile("s_wait_asynccnt %0" :: "n"(CNT) : "memory");
#endif
}

// ---------------------------------------------------------------------------
// Conversion / staging kernels (f32 -> bf16, transpose + zero-pad)
// ---------------------------------------------------------------------------

// keysT[m][d] = keys_bank[d][m] * 1/sqrt(128), zero for m >= BANK_N
__global__ void conv_keysT(const float* __restrict__ keys, __bf16* __restrict__ keysT) {
    const float scale = 0.08838834764831845f; // 1/sqrt(128)
    size_t idx = (size_t)blockIdx.x * 256 + threadIdx.x;
    if (idx >= (size_t)BANK_P * D_KEY) return;
    int m = (int)(idx >> 7);
    int d = (int)(idx & 127);
    float v = (m < BANK_N) ? keys[(size_t)d * BANK_N + m] * scale : 0.0f;
    keysT[idx] = (__bf16)v;
}

// qT[n][d] = q_in[d][n], zero for n >= N_COL
__global__ void conv_qT(const float* __restrict__ q_in, __bf16* __restrict__ qT) {
    size_t idx = (size_t)blockIdx.x * 256 + threadIdx.x;
    if (idx >= (size_t)N_PAD * D_KEY) return;
    int n = (int)(idx >> 7);
    int d = (int)(idx & 127);
    float v = (n < N_COL) ? q_in[(size_t)d * N_COL + n] : 0.0f;
    qT[idx] = (__bf16)v;
}

// vbf[r][k] = values[r][k] (r = o*512+m), zero for k >= BANK_N
__global__ void conv_values(const float* __restrict__ values, __bf16* __restrict__ vbf) {
    size_t idx = (size_t)blockIdx.x * 256 + threadIdx.x;
    if (idx >= (size_t)OBJ_N * D_VAL * BANK_P) return;
    size_t r = idx / BANK_P;
    int   k  = (int)(idx % BANK_P);
    float v = (k < BANK_N) ? values[r * BANK_N + k] : 0.0f;
    vbf[idx] = (__bf16)v;
}

// ---------------------------------------------------------------------------
// NT GEMM, bf16 WMMA:  C[m][n] = sum_k A[m][k] * Bt[n][k]   (f32 accumulate)
// A: M x Kp row-major bf16 (per-batch strideA), Bt: Npad x Kp row-major bf16.
// Block = 256 threads (8 waves), block tile 128x128, wave tile 64x32,
// K-step 32  ->  8x v_wmma_f32_16x16x32_bf16 per wave per K-step.
//
// Tile movement: GLOBAL_LOAD_ASYNC_TO_LDS_B128 into a 3-deep LDS pipeline
// (issue-ahead-1).  One barrier per K-step; s_wait_asynccnt<=4 keeps the
// next tile's DMA in flight while guaranteeing the current tile landed
// (ASYNCcnt completes in order).  Race-freedom: buffer written by
// issue(kt+1) (after barrier(kt-1)) was last read by compute(kt-2), which
// all waves finished before barrier(kt-1).  Buffer indices rotate via a
// conditional increment; LDS offsets are base + (buf << 13) arithmetic
// (buffers are 8 KB apart by construction), so no select chains.
// ---------------------------------------------------------------------------
__global__ __launch_bounds__(256) void gemm_nt_bf16(
    const __bf16* __restrict__ A, size_t strideA,
    const __bf16* __restrict__ Bt,
    float* __restrict__ Cp, size_t strideC,
    int M, int Nreal, int Kp, int ldc)
{
    __shared__ u32x4 sA[3][512];   // 3 x (128 rows x 32 bf16) = 24 KB, 8 KB apart
    __shared__ u32x4 sB[3][512];   // 3 x (128 rows x 32 bf16) = 24 KB, 8 KB apart

    const int tid   = threadIdx.x;
    const int lane  = tid & 31;
    const int wid   = tid >> 5;
    const int waveM = wid >> 2;            // 0..1  -> 64-row slab
    const int waveN = wid & 3;             // 0..3  -> 32-col slab
    const int m0    = blockIdx.y * 128;
    const int n0    = blockIdx.x * 128;
    const int b     = blockIdx.z;

    const __bf16* Ab = A  + (size_t)b * strideA;
    float*        Cb = Cp + (size_t)b * strideC;

    const int rowA = tid >> 2;             // 0..63
    const int colc = tid & 3;              // 16B chunk within 64B row

    // per-thread global source pointers (advance by k along the row)
    const __bf16* gA0 = Ab + (size_t)(m0 + rowA)      * Kp + colc * 8;
    const __bf16* gA1 = Ab + (size_t)(m0 + rowA + 64) * Kp + colc * 8;
    const __bf16* gB0 = Bt + (size_t)(n0 + rowA)      * Kp + colc * 8;
    const __bf16* gB1 = Bt + (size_t)(n0 + rowA + 64) * Kp + colc * 8;

    const int nk = Kp >> 5;

    // buffer-0 LDS byte offsets; buffer q is exactly q*8192 bytes further
    const unsigned oA = lds_off(&sA[0][tid]);
    const unsigned oB = lds_off(&sB[0][tid]);

    // issue one 32-K tile (4 async b128 per thread) into LDS buffer `buf`
    auto issue = [&](int kt, int buf) {
        const int      k  = kt << 5;
        const unsigned sh = (unsigned)buf << 13;     // 8 KB per buffer
        async_ld_b128(oA + sh,        gA0 + k);
        async_ld_b128(oA + sh + 4096, gA1 + k);      // chunk tid+256
        async_ld_b128(oB + sh,        gB0 + k);
        async_ld_b128(oB + sh + 4096, gB1 + k);
    };

    const int sel  = lane >> 4;            // K-half select per WMMA layout
    const int lrow = lane & 15;

    v8f acc[4][2];
    v8f zero = {};
    #pragma unroll
    for (int i = 0; i < 4; i++)
        #pragma unroll
        for (int j = 0; j < 2; j++) acc[i][j] = zero;

    issue(0, 0);                           // prologue: tile 0 in flight

    int cur = 0, nxt = 1;                  // rotating buffer indices
    for (int kt = 0; kt < nk; ++kt) {
        if (kt + 1 < nk) {
            issue(kt + 1, nxt);            // keep next tile's DMA in flight
            wait_asynccnt<4>();            // current tile's 4 DMAs complete
        } else {
            wait_asynccnt<0>();            // drain
        }
        __syncthreads();                   // tile `cur` visible to all waves

        // ---- fragment assembly per ISA 16-bit A(16x32)/B(32x16) layout ----
        // A: lane<16 holds K 0..7 & 16..23, lane>=16 holds K 8..15 & 24..31.
        FragCvt afr[4], bfr[2];
        #pragma unroll
        for (int ni = 0; ni < 2; ni++) {
            int r = waveN * 32 + ni * 16 + lrow;
            bfr[ni].u[0] = sB[cur][r * 4 + sel];
            bfr[ni].u[1] = sB[cur][r * 4 + 2 + sel];
        }
        #pragma unroll
        for (int mi = 0; mi < 4; mi++) {
            int r = waveM * 64 + mi * 16 + lrow;
            afr[mi].u[0] = sA[cur][r * 4 + sel];        // K 0..7   / 8..15
            afr[mi].u[1] = sA[cur][r * 4 + 2 + sel];    // K 16..23 / 24..31
        }

        #pragma unroll
        for (int mi = 0; mi < 4; mi++)
            #pragma unroll
            for (int ni = 0; ni < 2; ni++)
                acc[mi][ni] = __builtin_amdgcn_wmma_f32_16x16x32_bf16(
                    false, afr[mi].v, false, bfr[ni].v,
                    (short)0, acc[mi][ni], false, false);

        cur = nxt;
        nxt = (nxt == 2) ? 0 : nxt + 1;    // cheap rotation, no division
    }

    // ---- store: C/D layout — VGPR v holds (M = sel*8 + v, N = lane&15) ----
    #pragma unroll
    for (int mi = 0; mi < 4; mi++) {
        #pragma unroll
        for (int ni = 0; ni < 2; ni++) {
            int n = n0 + waveN * 32 + ni * 16 + lrow;
            #pragma unroll
            for (int v = 0; v < 8; v++) {
                int m = m0 + waveM * 64 + mi * 16 + sel * 8 + v;
                if (m < M && n < Nreal)
                    Cb[(size_t)m * ldc + n] = acc[mi][ni][v];
            }
        }
    }
}

// ---------------------------------------------------------------------------
// Column softmax over bank dim; writes pT[n][k] in bf16 (K- and N-padded).
// Thread per column; coalesced reads of S (ld = N_PAD).
// ---------------------------------------------------------------------------
__global__ __launch_bounds__(256) void softmax_cols(
    const float* __restrict__ S, __bf16* __restrict__ pT)
{
    int n = blockIdx.x * 256 + threadIdx.x;
    if (n >= N_PAD) return;
    __bf16* row = pT + (size_t)n * BANK_P;
    if (n >= N_COL) {                      // padding columns: all-zero p
        for (int k = 0; k < BANK_P; k++) row[k] = (__bf16)0.0f;
        return;
    }
    float m = -3.402823466e38f, s = 0.0f;
    for (int k = 0; k < BANK_N; k++) {     // online softmax (single pass stats)
        float x  = S[(size_t)k * N_PAD + n];
        float mn = fmaxf(m, x);
        s = s * __expf(m - mn) + __expf(x - mn);
        m = mn;
    }
    float inv = 1.0f / s;
    for (int k = 0; k < BANK_N; k++) {
        float x = S[(size_t)k * N_PAD + n];
        row[k] = (__bf16)(__expf(x - m) * inv);
    }
    for (int k = BANK_N; k < BANK_P; k++) row[k] = (__bf16)0.0f;
}

// ---------------------------------------------------------------------------
// mask_mem[o][n] = sum_k masks[o][k] * p[k][n]  (pT row n is contiguous).
// masks row staged in LDS (38.9 KB of the 320 KB WGP pool).
// ---------------------------------------------------------------------------
__global__ __launch_bounds__(256) void mask_dot(
    const __bf16* __restrict__ pT, const float* __restrict__ masks,
    float* __restrict__ mm)
{
    __shared__ float smask[BANK_N];
    int o = blockIdx.y;
    for (int k = threadIdx.x; k < BANK_N; k += 256)
        smask[k] = masks[(size_t)o * BANK_N + k];
    __syncthreads();

    int n = blockIdx.x * 256 + threadIdx.x;
    if (n >= N_COL) return;
    const __bf16* row = pT + (size_t)n * BANK_P;
    float acc = 0.0f;
    for (int k = 0; k < BANK_N; k += 8) {  // 9720 % 8 == 0
        Pack8 p8; p8.u = *(const u32x4*)(row + k);
        #pragma unroll
        for (int j = 0; j < 8; j++) acc += smask[k + j] * (float)p8.h[j];
    }
    mm[(size_t)o * N_PAD + n] = acc;
}

// ---------------------------------------------------------------------------
// out[o][512+c][n] = q_out[o][c][n] * mask_mem[o][n]
// ---------------------------------------------------------------------------
__global__ void fuse_qout(const float* __restrict__ q_out,
                          const float* __restrict__ mm,
                          float* __restrict__ out)
{
    size_t idx = (size_t)blockIdx.x * 256 + threadIdx.x;
    const size_t total = (size_t)OBJ_N * C_CH * N_COL;
    if (idx >= total) return;
    int    n = (int)(idx % N_COL);
    size_t t = idx / N_COL;
    int    c = (int)(t % C_CH);
    int    o = (int)(t / C_CH);
    out[((size_t)o * (D_VAL + C_CH) + D_VAL + c) * N_COL + n] =
        q_out[idx] * mm[(size_t)o * N_PAD + n];
}

// ---------------------------------------------------------------------------
// Host launcher
// ---------------------------------------------------------------------------
extern "C" void kernel_launch(void* const* d_in, const int* in_sizes, int n_in,
                              void* d_out, int out_size, void* d_ws, size_t ws_size,
                              hipStream_t stream)
{
    (void)in_sizes; (void)n_in; (void)out_size;

    const float* keys   = (const float*)d_in[0];  // (128, 9720)
    const float* q_in   = (const float*)d_in[1];  // (1, 128, 1620)
    const float* q_out  = (const float*)d_in[2];  // (3, 512, 1620)
    const float* values = (const float*)d_in[3];  // (3, 512, 9720)
    const float* masks  = (const float*)d_in[4];  // (3, 1, 9720)
    float* out = (float*)d_out;                   // (1, 3, 1024, 1620)

    // ---- workspace layout (all 256B-aligned sizes) ----
    const size_t SZ_KEYST = (size_t)BANK_P * D_KEY * 2;            //  2.49 MB
    const size_t SZ_QT    = (size_t)N_PAD  * D_KEY * 2;            //  0.43 MB
    const size_t SZ_VBF   = (size_t)OBJ_N * D_VAL * BANK_P * 2;    // 29.9  MB
    const size_t SZ_S     = (size_t)BANK_P * N_PAD * 4;            // 64.7  MB
    const size_t SZ_PT    = (size_t)N_PAD * BANK_P * 2;            // 32.4  MB
    const size_t SZ_MM    = (size_t)OBJ_N * N_PAD * 4;             // 20 KB
    const size_t NEEDED = SZ_KEYST + SZ_QT + SZ_VBF + SZ_S + SZ_PT + SZ_MM;
    if (ws_size < NEEDED) return;  // cannot allocate during graph capture

    char* ws = (char*)d_ws;
    __bf16* keysT = (__bf16*)(ws);                     ws += SZ_KEYST;
    __bf16* qT    = (__bf16*)(ws);                     ws += SZ_QT;
    __bf16* vbf   = (__bf16*)(ws);                     ws += SZ_VBF;
    float*  S     = (float*) (ws);                     ws += SZ_S;
    __bf16* pT    = (__bf16*)(ws);                     ws += SZ_PT;
    float*  mm    = (float*) (ws);

    // 1) stage bf16 operands (transpose keys/q, bake softmax scale into keys)
    conv_keysT <<<(BANK_P * D_KEY + 255) / 256, 256, 0, stream>>>(keys, keysT);
    conv_qT    <<<(N_PAD  * D_KEY + 255) / 256, 256, 0, stream>>>(q_in, qT);
    conv_values<<<((size_t)OBJ_N * D_VAL * BANK_P + 255) / 256, 256, 0, stream>>>(values, vbf);

    // 2) logits: S[bank][n] = keysT x qT^T   (M=9728, N=1664, K=128)
    gemm_nt_bf16<<<dim3(N_PAD / 128, BANK_P / 128, 1), 256, 0, stream>>>(
        keysT, 0, qT, S, 0, BANK_P, N_PAD, D_KEY, N_PAD);

    // 3) softmax over bank dim -> pT (bf16, transposed, padded)
    softmax_cols<<<(N_PAD + 255) / 256, 256, 0, stream>>>(S, pT);

    // 4) mem[o] = values[o] x p  -> first 512 channels of d_out
    //    (M=512, N=1620, K=9728, batched over 3 objects)
    gemm_nt_bf16<<<dim3(N_PAD / 128, D_VAL / 128, OBJ_N), 256, 0, stream>>>(
        vbf, (size_t)D_VAL * BANK_P, pT,
        out, (size_t)(D_VAL + C_CH) * N_COL,
        D_VAL, N_COL, BANK_P, N_COL);

    // 5) mask_mem[o][n] = masks[o] . p[:,n]
    mask_dot<<<dim3((N_COL + 255) / 256, OBJ_N), 256, 0, stream>>>(pT, masks, mm);

    // 6) out[o][512+c][n] = q_out[o][c][n] * mask_mem[o][n]
    fuse_qout<<<((size_t)OBJ_N * C_CH * N_COL + 255) / 256, 256, 0, stream>>>(q_out, mm, out);
}
